// AgnosticNMS_14001593384971
// MI455X (gfx1250) — compile-verified, matrix operand order checked by reading, not run on hardware
//
#include <hip/hip_runtime.h>

// Problem constants from the reference
#define NB    8
#define NBOX  25200
#define NCLS  80
#define TOPK  100
#define CONF_T 0.25f
#define IOU_T  0.45f

typedef __attribute__((ext_vector_type(16))) _Float16 v16h;
typedef __attribute__((ext_vector_type(8)))  float    v8f;

// ---------------------------------------------------------------------------
// Kernel 1: per-box reduction over the class dimension (memory-bound, full
// grid). Each thread consumes two contiguous 320B rows (classes + scores)
// via float4 loads -> whole cachelines per lane.
// ---------------------------------------------------------------------------
__global__ __launch_bounds__(256) void nms_preprocess(
    const float* __restrict__ classes, const float* __restrict__ scores,
    float* __restrict__ ws_score, float* __restrict__ ws_cls) {
  int r = blockIdx.x * blockDim.x + threadIdx.x;
  if (r >= NB * NBOX) return;
  const float4* cs = (const float4*)(classes + (size_t)r * NCLS);
  const float4* ss = (const float4*)(scores  + (size_t)r * NCLS);
  float cbest = -INFINITY; int cidx = 0;
  float sbest = -INFINITY;
#pragma unroll
  for (int j = 0; j < NCLS / 4; ++j) {
    float4 c = cs[j];
    float4 s = ss[j];
    if (c.x > cbest) { cbest = c.x; cidx = 4 * j + 0; }
    if (c.y > cbest) { cbest = c.y; cidx = 4 * j + 1; }
    if (c.z > cbest) { cbest = c.z; cidx = 4 * j + 2; }
    if (c.w > cbest) { cbest = c.w; cidx = 4 * j + 3; }
    sbest = fmaxf(sbest, fmaxf(fmaxf(s.x, s.y), fmaxf(s.z, s.w)));
  }
  ws_score[r] = sbest;
  ws_cls[r]   = (float)cidx;
}

// ---------------------------------------------------------------------------
// Kernel 2: sequential NMS, one workgroup (1024 threads = 32 wave32) per
// batch element. Scores live in LDS (100.8 KB of the 320 KB WGP budget),
// staged with async global->LDS B128 copies. All LDS sweeps use B128
// accesses. Boxes stream from L2 each iteration. Final valid-count uses one
// v_wmma against a ones matrix.
// ---------------------------------------------------------------------------
__global__ __launch_bounds__(1024) void nms_iter(
    const float* __restrict__ boxes,      // [B, N, 4] (y1,x1,y2,x2)
    const float* __restrict__ ws_score,   // [B, N]
    const float* __restrict__ ws_cls,     // [B, N]
    float* __restrict__ out_boxes,        // [B, TOPK, 4]
    float* __restrict__ out_scores,       // [B, TOPK]
    float* __restrict__ out_cls,          // [B, TOPK]
    float* __restrict__ out_valid) {      // [B]
  __shared__ float    s_score[NBOX];   // 100800 B, 16B aligned
  __shared__ float    s_rv[32];
  __shared__ int      s_ri[32];
  __shared__ float    s_b[6];          // y1,x1,y2,x2,area,keep
  __shared__ _Float16 s_kept[512];     // kept flags padded to a full A-matrix

  const int tid  = threadIdx.x;
  const int b    = blockIdx.x;
  const int lane = tid & 31;
  const int wid  = tid >> 5;

  if (tid < 512) s_kept[tid] = (_Float16)0.0f;

  // ---- async stage: ws_score[b, :] -> LDS (B128 per lane, ASYNCcnt) ----
  const float* src = ws_score + (size_t)b * NBOX;
  for (int q = tid; q < NBOX / 4; q += 1024) {
    unsigned     loff = (unsigned)(uintptr_t)(&s_score[q * 4]);
    const float* g    = src + q * 4;
    asm volatile("global_load_async_to_lds_b128 %0, %1, off"
                 :: "v"(loff), "v"(g) : "memory");
  }
  asm volatile("s_wait_asynccnt 0" ::: "memory");
  __syncthreads();

  const float4* bx4 = (const float4*)boxes + (size_t)b * NBOX;
  const float4* s4  = (const float4*)s_score;
  float4*       s4w = (float4*)s_score;

  for (int k = 0; k < TOPK; ++k) {
    // ---- parallel argmax over LDS scores (ds_load_b128 sweep) ----
    float bv = -INFINITY;
    int   bi = 0x7fffffff;
    for (int q = tid; q < NBOX / 4; q += 1024) {
      float4 sc  = s4[q];
      int    bse = q * 4;
      if (sc.x > bv || (sc.x == bv && bse + 0 < bi)) { bv = sc.x; bi = bse + 0; }
      if (sc.y > bv || (sc.y == bv && bse + 1 < bi)) { bv = sc.y; bi = bse + 1; }
      if (sc.z > bv || (sc.z == bv && bse + 2 < bi)) { bv = sc.z; bi = bse + 2; }
      if (sc.w > bv || (sc.w == bv && bse + 3 < bi)) { bv = sc.w; bi = bse + 3; }
    }
#pragma unroll
    for (int off = 16; off > 0; off >>= 1) {
      float ov = __shfl_down(bv, (unsigned)off, 32);
      int   oi = __shfl_down(bi, (unsigned)off, 32);
      if (ov > bv || (ov == bv && oi < bi)) { bv = ov; bi = oi; }
    }
    if (lane == 0) { s_rv[wid] = bv; s_ri[wid] = bi; }
    __syncthreads();
    if (wid == 0) {
      bv = s_rv[lane];
      bi = s_ri[lane];
#pragma unroll
      for (int off = 16; off > 0; off >>= 1) {
        float ov = __shfl_down(bv, (unsigned)off, 32);
        int   oi = __shfl_down(bi, (unsigned)off, 32);
        if (ov > bv || (ov == bv && oi < bi)) { bv = ov; bi = oi; }
      }
      if (lane == 0) {
        bool   keep = bv >= CONF_T;
        float4 bb   = bx4[bi];
        float  area = (bb.z - bb.x) * (bb.w - bb.y);
        s_b[0] = bb.x; s_b[1] = bb.y; s_b[2] = bb.z; s_b[3] = bb.w;
        s_b[4] = area; s_b[5] = keep ? 1.0f : 0.0f;
        s_score[bi] = -INFINITY;            // always remove the selected box
        s_kept[k]   = keep ? (_Float16)1.0f : (_Float16)0.0f;
        out_scores[b * TOPK + k] = keep ? bv : -1.0f;
        out_cls[b * TOPK + k] =
            keep ? ws_cls[(size_t)b * NBOX + bi] : -1.0f;
        float4* ob4 = (float4*)out_boxes;
        ob4[b * TOPK + k] = keep ? bb : make_float4(0.f, 0.f, 0.f, 0.f);
      }
    }
    __syncthreads();

    // ---- suppression sweep (ds_load_b128 liveness check, owner-writeback) --
    if (s_b[5] > 0.0f) {
      const float y1s = s_b[0], x1s = s_b[1], y2s = s_b[2], x2s = s_b[3];
      const float as  = s_b[4];
      for (int q = tid; q < NBOX / 4; q += 1024) {
        float4 sc = s4[q];
        float  vs[4] = {sc.x, sc.y, sc.z, sc.w};
        bool   any = false;
#pragma unroll
        for (int e = 0; e < 4; ++e) {
          if (vs[e] == -INFINITY) continue;     // dead: skip L2 read
          float4 bb   = bx4[q * 4 + e];
          float iy1   = fmaxf(bb.x, y1s);
          float ix1   = fmaxf(bb.y, x1s);
          float iy2   = fminf(bb.z, y2s);
          float ix2   = fminf(bb.w, x2s);
          float inter = fmaxf(iy2 - iy1, 0.0f) * fmaxf(ix2 - ix1, 0.0f);
          float ar    = (bb.z - bb.x) * (bb.w - bb.y);
          float uni   = fmaxf(ar + as - inter, 1e-9f);
          if (inter / uni > IOU_T) { vs[e] = -INFINITY; any = true; }
        }
        if (any) {
          s4w[q] = make_float4(vs[0], vs[1], vs[2], vs[3]);
        }
      }
    }
    __syncthreads();
  }

  // ---- valid = sum(kept) via one WMMA (B = ones -> row sums in D) ----
  if (tid < 32) {   // wave 0 only; EXEC is all-ones within the wave
    v16h a, on;
#pragma unroll
    for (int j = 0; j < 16; ++j) {
      a[j]  = s_kept[lane * 16 + j];
      on[j] = (_Float16)1.0f;
    }
    v8f acc = {};
    acc = __builtin_amdgcn_wmma_f32_16x16x32_f16(
        false, a, false, on, (short)0, acc, false, false);
    float ssum = acc[0] + acc[1] + acc[2] + acc[3] +
                 acc[4] + acc[5] + acc[6] + acc[7];
    // lane 0 holds rows 0..7 (n=0), lane 16 holds rows 8..15 (n=0)
    float other = __shfl_down(ssum, 16u, 32);
    if (lane == 0) out_valid[b] = ssum + other;
  }
}

// ---------------------------------------------------------------------------
extern "C" void kernel_launch(void* const* d_in, const int* in_sizes, int n_in,
                              void* d_out, int out_size, void* d_ws,
                              size_t ws_size, hipStream_t stream) {
  const float* boxes   = (const float*)d_in[0];
  const float* classes = (const float*)d_in[1];
  const float* scores  = (const float*)d_in[2];
  // d_in[3] = topk_all (fixed at 100 by the reference setup)

  float* ws_score = (float*)d_ws;              // B*N floats
  float* ws_cls   = ws_score + NB * NBOX;      // B*N floats

  float* ob = (float*)d_out;                   // [B,100,4]
  float* os = ob + NB * TOPK * 4;              // [B,100]
  float* oc = os + NB * TOPK;                  // [B,100]
  float* ov = oc + NB * TOPK;                  // [B]

  const int rows = NB * NBOX;
  nms_preprocess<<<(rows + 255) / 256, 256, 0, stream>>>(
      classes, scores, ws_score, ws_cls);
  nms_iter<<<NB, 1024, 0, stream>>>(
      boxes, ws_score, ws_cls, ob, os, oc, ov);
}
